// PairMixing_79886391705995
// MI455X (gfx1250) — matrix operand order repaired and checked by reading, not compiled, not run on hardware
//
#include <hip/hip_runtime.h>

// Pair mixing (equivariant CG tensor product), MI455X / gfx1250, wave32.
// N=60000 edges, F=128 features, B=32 basis, ORDER=2 -> 15 (l1,l2,L) triples.
//
// Strategy:
//  * coeff_t = rbf @ W_t^T via V_WMMA_F32_16X16X32_BF16 (K=32 == B, one WMMA
//    per 16-edge x 16-feature tile per triple), f32 accumulate.
//  * CG contraction is elementwise in f -> f32 VALU FMAs. Refactored as
//    tp[k] = sum_ij cg[i,j,k]*(x1_i*x2_j), then acc += cf*tp: ~871 ops/slot
//    instead of 1025 (coeff applied once per output component, not per cg
//    entry), and the FMA block no longer depends on the WMMA result.
//  * Fully fused: no coeff intermediate in HBM (saves ~920 MB of traffic).
//    HBM floor: x1+x2+y ~= 830 MB -> ~36 us at 23.3 TB/s.
//  * Cache policy: x/y are streamed once -> non-temporal loads/stores; W/cg/rbf
//    are re-read by all blocks -> regular temporal so they stay in the 192 MB L2.

#define N_EDGES 60000
#define NF      128
#define NB      32

typedef __bf16 v16bf __attribute__((ext_vector_type(16)));
typedef float  v8f   __attribute__((ext_vector_type(8)));

struct Params {
  const float* x1[3];     // x1_l : [N, 2l+1, F]
  const float* x2[3];     // x2_l : [N, 2l+1, F]
  const float* rbf;       // [N, B]
  const float* cg[15];    // [2l1+1, 2l2+1, 2L+1]
  const float* W[15];     // [F, B]
  float* y0;              // [N, 1, F]
  float* y1;              // [N, 3, F]
  float* y2;              // [N, 5, F]
};

__global__ __launch_bounds__(256) void pair_mixing_kernel(Params p) {
  constexpr int TL1[15] = {0,0,0,1,1,1,1,1,1,2,2,2,2,2,2};
  constexpr int TL2[15] = {0,1,2,0,1,1,1,2,2,0,1,1,2,2,2};
  constexpr int TLo[15] = {0,1,2,1,0,1,2,1,2,2,1,2,0,1,2};

  const int lane  = threadIdx.x & 31;
  const int wave  = threadIdx.x >> 5;
  const int lo    = lane & 15;
  const int hi    = lane >> 4;
  const int nbase = blockIdx.x * 16;   // 16 edges per block
  const int fbase = wave * 16;         // 16 features per wave
  const int f     = fbase + lo;

  // Speculative prefetch of the next edge tile's x data (global_prefetch_b8).
  if (blockIdx.x + 1 < gridDim.x) {
    const int pn = nbase + 16;
    __builtin_prefetch(p.x1[0] + (size_t)pn * NF       + threadIdx.x, 0, 1);
    __builtin_prefetch(p.x1[1] + (size_t)pn * 3 * NF   + threadIdx.x, 0, 1);
    __builtin_prefetch(p.x1[2] + (size_t)pn * 5 * NF   + threadIdx.x, 0, 1);
    __builtin_prefetch(p.x2[0] + (size_t)pn * NF       + threadIdx.x, 0, 1);
    __builtin_prefetch(p.x2[1] + (size_t)pn * 3 * NF   + threadIdx.x, 0, 1);
    __builtin_prefetch(p.x2[2] + (size_t)pn * 5 * NF   + threadIdx.x, 0, 1);
  }

  // ---- A matrix: rbf tile [16 edges x 32 basis], bf16, dense 16-bit A layout.
  // Lanes 0-15 (hi=0): M=lane, K elements {0..7, 16..23}
  // Lanes 16-31 (hi=1): M=lane-16, K elements {8..15, 24..31}
  v16bf A;
  {
    const float* rrow = p.rbf + (size_t)(nbase + lo) * NB;
    const int k0 = hi ? 8 : 0;
    const int k1 = hi ? 24 : 16;
    const float4* r4 = reinterpret_cast<const float4*>(rrow);
    float4 q0 = r4[(k0 >> 2) + 0];
    float4 q1 = r4[(k0 >> 2) + 1];
    float4 q2 = r4[(k1 >> 2) + 0];
    float4 q3 = r4[(k1 >> 2) + 1];
    A[0]=(__bf16)q0.x;  A[1]=(__bf16)q0.y;  A[2]=(__bf16)q0.z;  A[3]=(__bf16)q0.w;
    A[4]=(__bf16)q1.x;  A[5]=(__bf16)q1.y;  A[6]=(__bf16)q1.z;  A[7]=(__bf16)q1.w;
    A[8]=(__bf16)q2.x;  A[9]=(__bf16)q2.y;  A[10]=(__bf16)q2.z; A[11]=(__bf16)q2.w;
    A[12]=(__bf16)q3.x; A[13]=(__bf16)q3.y; A[14]=(__bf16)q3.z; A[15]=(__bf16)q3.w;
  }

  // ---- Load x1/x2 for this lane's 8 (edge, f) slots; reused by all triples.
  // WMMA C layout: VGPR r, lanes 0-15 -> edge r, lanes 16-31 -> edge 8+r.
  // Streamed once -> non-temporal (keep W/cg hot in L2 instead).
  float x1v[8][9];
  float x2v[8][9];
  #pragma unroll
  for (int r = 0; r < 8; ++r) {
    const size_t n = (size_t)(nbase + r + 8 * hi);
    x1v[r][0] = __builtin_nontemporal_load(p.x1[0] + n * NF + f);
    x2v[r][0] = __builtin_nontemporal_load(p.x2[0] + n * NF + f);
    #pragma unroll
    for (int c = 0; c < 3; ++c) {
      x1v[r][1 + c] = __builtin_nontemporal_load(p.x1[1] + (n * 3 + c) * NF + f);
      x2v[r][1 + c] = __builtin_nontemporal_load(p.x2[1] + (n * 3 + c) * NF + f);
    }
    #pragma unroll
    for (int c = 0; c < 5; ++c) {
      x1v[r][4 + c] = __builtin_nontemporal_load(p.x1[2] + (n * 5 + c) * NF + f);
      x2v[r][4 + c] = __builtin_nontemporal_load(p.x2[2] + (n * 5 + c) * NF + f);
    }
  }

  // ---- Output accumulators: 9 components (L=0:1, L=1:3, L=2:5) per slot.
  float acc[8][9];
  #pragma unroll
  for (int r = 0; r < 8; ++r)
    #pragma unroll
    for (int k = 0; k < 9; ++k)
      acc[r][k] = 0.0f;

  // ---- Per-triple: WMMA coeff tile + CG tensor product, combined at the end.
  #pragma unroll
  for (int t = 0; t < 15; ++t) {
    const int l1 = TL1[t], l2 = TL2[t], L = TLo[t];
    const int d1 = 2 * l1 + 1, d2 = 2 * l2 + 1, dL = 2 * L + 1;
    const int o1 = l1 * l1, o2 = l2 * l2, oL = L * L;

    // B matrix: W_t^T slice [32 basis x 16 features], bf16.
    // Table layout (matches A): lanes 0-15 hold K={0..7,16..23} of column n=lane,
    // lanes 16-31 hold K={8..15,24..31} of column n=lane-16.
    v16bf Bm;
    {
      const float* wrow = p.W[t] + (size_t)f * NB;  // W[f][0..31]
      const int kb0 = hi * 8;
      const int kb1 = 16 + hi * 8;
      const float4* w4 = reinterpret_cast<const float4*>(wrow);
      float4 q0 = w4[(kb0 >> 2) + 0];
      float4 q1 = w4[(kb0 >> 2) + 1];
      float4 q2 = w4[(kb1 >> 2) + 0];
      float4 q3 = w4[(kb1 >> 2) + 1];
      Bm[0]=(__bf16)q0.x;  Bm[1]=(__bf16)q0.y;  Bm[2]=(__bf16)q0.z;  Bm[3]=(__bf16)q0.w;
      Bm[4]=(__bf16)q1.x;  Bm[5]=(__bf16)q1.y;  Bm[6]=(__bf16)q1.z;  Bm[7]=(__bf16)q1.w;
      Bm[8]=(__bf16)q2.x;  Bm[9]=(__bf16)q2.y;  Bm[10]=(__bf16)q2.z; Bm[11]=(__bf16)q2.w;
      Bm[12]=(__bf16)q3.x; Bm[13]=(__bf16)q3.y; Bm[14]=(__bf16)q3.z; Bm[15]=(__bf16)q3.w;
    }

    v8f cz = {};
    // coeff[edge, f] = sum_b rbf[edge,b] * W_t[f,b]
    v8f cf = __builtin_amdgcn_wmma_f32_16x16x32_bf16(
        /*neg_a=*/false, A, /*neg_b=*/false, Bm,
        /*c_mod=*/(short)0, cz, /*reuse_a=*/false, /*reuse_b=*/false);

    // Raw tensor product tp[r][k] = sum_ij cg[i,j,k] * x1_i * x2_j.
    // Independent of the WMMA result -> hides WMMA latency under VALU work.
    float tp[8][5];
    #pragma unroll
    for (int r = 0; r < 8; ++r)
      #pragma unroll
      for (int k = 0; k < dL; ++k)
        tp[r][k] = 0.0f;

    const float* cgp = p.cg[t];   // uniform across the wave -> scalar loads
    #pragma unroll
    for (int i = 0; i < d1; ++i) {
      #pragma unroll
      for (int j = 0; j < d2; ++j) {
        float cgk[5];
        #pragma unroll
        for (int k = 0; k < dL; ++k)
          cgk[k] = cgp[(i * d2 + j) * dL + k];
        #pragma unroll
        for (int r = 0; r < 8; ++r) {
          const float prod = x1v[r][o1 + i] * x2v[r][o2 + j];
          #pragma unroll
          for (int k = 0; k < dL; ++k)
            tp[r][k] = fmaf(cgk[k], prod, tp[r][k]);
        }
      }
    }

    // acc += coeff * tp  (coeff applied once per output component).
    #pragma unroll
    for (int r = 0; r < 8; ++r)
      #pragma unroll
      for (int k = 0; k < dL; ++k)
        acc[r][oL + k] = fmaf(cf[r], tp[r][k], acc[r][oL + k]);
  }

  // ---- Store y0/y1/y2 (each output element owned by exactly one slot).
  // Streamed once -> non-temporal stores.
  #pragma unroll
  for (int r = 0; r < 8; ++r) {
    const size_t n = (size_t)(nbase + r + 8 * hi);
    __builtin_nontemporal_store(acc[r][0], p.y0 + n * NF + f);
    #pragma unroll
    for (int c = 0; c < 3; ++c)
      __builtin_nontemporal_store(acc[r][1 + c], p.y1 + (n * 3 + c) * NF + f);
    #pragma unroll
    for (int c = 0; c < 5; ++c)
      __builtin_nontemporal_store(acc[r][4 + c], p.y2 + (n * 5 + c) * NF + f);
  }
}

extern "C" void kernel_launch(void* const* d_in, const int* in_sizes, int n_in,
                              void* d_out, int out_size, void* d_ws, size_t ws_size,
                              hipStream_t stream) {
  (void)in_sizes; (void)n_in; (void)out_size; (void)d_ws; (void)ws_size;
  Params p;
  p.x1[0] = (const float*)d_in[0];
  p.x1[1] = (const float*)d_in[1];
  p.x1[2] = (const float*)d_in[2];
  p.x2[0] = (const float*)d_in[3];
  p.x2[1] = (const float*)d_in[4];
  p.x2[2] = (const float*)d_in[5];
  p.rbf   = (const float*)d_in[6];
  // setup_inputs() interleaves cg_t / W_t per triple.
  for (int i = 0; i < 15; ++i) {
    p.cg[i] = (const float*)d_in[7 + 2 * i];
    p.W[i]  = (const float*)d_in[8 + 2 * i];
  }
  float* out = (float*)d_out;
  p.y0 = out;
  p.y1 = out + (size_t)N_EDGES * NF;        // after [N,1,F]
  p.y2 = out + (size_t)4 * N_EDGES * NF;    // after [N,1,F]+[N,3,F]

  dim3 grid(N_EDGES / 16);   // 3750 edge tiles
  dim3 block(256);           // 8 waves; wave w owns features [16w, 16w+16)
  pair_mixing_kernel<<<grid, block, 0, stream>>>(p);
}
